// myConv_subNetwork_53051436040855
// MI455X (gfx1250) — compile-verified, matrix operand order checked by reading, not run on hardware
//
#include <hip/hip_runtime.h>
#include <hip/hip_bf16.h>

typedef __attribute__((ext_vector_type(16))) _Float16 v16h;
typedef __attribute__((ext_vector_type(8)))  float    v8f;

#define HW        512
#define PLANE     (512 * 512)             // 262144 floats per (n,c) plane
#define HEAD_SIZE (8 * 3 * PLANE)         // 6291456: one (N,3,H,W) output tensor
#define LIGHT_OFF (3 * HEAD_SIZE)         // 18874368
#define LOSS_IDX  (LIGHT_OFF + 8 * 27 * 7 * 7)  // 18884952

// ---- native CDNA5 hardware tanh (TRANS32 op) -------------------------------
__device__ __forceinline__ float fast_tanh(float x) {
#if __has_builtin(__builtin_amdgcn_tanhf)
    return __builtin_amdgcn_tanhf(x);
#else
    float r;
    // v_nop covers the TRANS-op result hazard (1 indep op before use)
    asm volatile("v_tanh_f32 %0, %1\n\tv_nop" : "=v"(r) : "v"(x));
    return r;
#endif
}

// ---- stitch weight (partition-of-unity ramps), separable per axis ----------
__device__ __forceinline__ float edge_w(int i, int p) {
    // replicate _patch_weight: up/left flag <=> i==0, down/right flag <=> i==6
    if (p < 64 && i != 0) return (float)p;
    if (p > 64 && i != 6) return (float)(128 - p);
    return 64.0f;
}
__device__ __forceinline__ float edge_sum(int u) {
    int i1 = u >> 6;               // covering patch with offset in [0,64)
    float s = 0.0f;
    if (i1 <= 6) s += edge_w(i1, u - (i1 << 6));
    if (i1 >= 1) s += edge_w(i1 - 1, u - ((i1 - 1) << 6));
    return s;
}

// ---- main fused kernel: 9x6 head matmul via WMMA + tanh + stitch -----------
__global__ __launch_bounds__(256)
void heads_wmma_kernel(const float* __restrict__ inT, const float* __restrict__ inN,
                       const float* __restrict__ Wa, const float* __restrict__ ba,
                       const float* __restrict__ Wn, const float* __restrict__ bn,
                       const float* __restrict__ Ws, const float* __restrict__ bs,
                       float* __restrict__ out)
{
    const int tid  = threadIdx.x;
    const int wave = tid >> 5;
    const int lane = tid & 31;

    // wave unit handles 64 consecutive pixels of one (n, y) row
    const int wid   = blockIdx.x * 8 + wave;   // [0, 32768)
    const int xq    = wid & 7;                 // 64-pixel eighth-of-row
    const int rowid = wid >> 3;                // n*512 + y
    const int y     = rowid & 511;
    const int n     = rowid >> 9;
    const int xbase = xq * 64;

    // A matrix (16x32 f16): rows 0..8 = stacked [W_albedo; W_normal; W_shading]
    // lane m<16 holds row m's K=0..7 halves in elements 0..7 (we use K=0..5)
    v16h a = {};
    if (lane < 9) {
        const float* Wm = (lane < 3) ? (Wa + lane * 6)
                        : (lane < 6) ? (Wn + (lane - 3) * 6)
                                     : (Ws + (lane - 6) * 6);
#pragma unroll
        for (int c = 0; c < 6; ++c) a[c] = (_Float16)Wm[c];
    }

    // bias template following the 16x16 f32 C/D layout (lanes 16-31 -> M=r+8)
    float biasr[8];
#pragma unroll
    for (int r = 0; r < 8; ++r) {
        int m = r + ((lane < 16) ? 0 : 8);
        biasr[r] = (m < 3) ? ba[m] : (m < 6) ? bn[m - 3] : (m < 9) ? bs[m - 6] : 0.0f;
    }

    const float  sumh    = edge_sum(y);
    const size_t imgbase = (size_t)(n * 3) * PLANE + (size_t)y * HW;

#pragma unroll
    for (int g = 0; g < 4; ++g) {
        const int x0 = xbase + g * 16;
        const int xp = x0 + (lane & 15);       // pixel column this lane owns

        // B matrix (32x16 f16): lane<16 holds column (pixel) xp, K=0..5 filled
        v16h b = {};
        if (lane < 16) {
            const size_t p0 = imgbase + xp;
            b[0] = (_Float16)inT[p0];
            b[1] = (_Float16)inT[p0 + PLANE];
            b[2] = (_Float16)inT[p0 + 2 * PLANE];
            b[3] = (_Float16)inN[p0];
            b[4] = (_Float16)inN[p0 + PLANE];
            b[5] = (_Float16)inN[p0 + 2 * PLANE];
        }

        v8f acc;
#pragma unroll
        for (int r = 0; r < 8; ++r) acc[r] = biasr[r];

        acc = __builtin_amdgcn_wmma_f32_16x16x32_f16(
                  false, a, false, b, (short)0, acc, false, false);

        // uniform post-processing: hardware tanh + stitch weight
        const float S = sumh * edge_sum(xp) * (1.0f / 4096.0f);
        float vals[8];
#pragma unroll
        for (int r = 0; r < 8; ++r) vals[r] = fast_tanh(acc[r]) * S;

        const size_t pb = imgbase + xp;
        if (lane < 16) {
            out[pb]                           = vals[0];  // albedo o0
            out[pb + PLANE]                   = vals[1];  // albedo o1
            out[pb + 2 * PLANE]               = vals[2];  // albedo o2
            out[HEAD_SIZE + pb]               = vals[3];  // normal o0
            out[HEAD_SIZE + pb + PLANE]       = vals[4];  // normal o1
            out[HEAD_SIZE + pb + 2 * PLANE]   = vals[5];  // normal o2
            out[2 * HEAD_SIZE + pb]           = vals[6];  // shading o0
            out[2 * HEAD_SIZE + pb + PLANE]   = vals[7];  // shading o1
        } else {
            out[2 * HEAD_SIZE + pb + 2 * PLANE] = vals[0]; // shading o2 (M=8)
        }
    }
}

// ---- 64x64 tile sums for the pooled means (LDS tree reduction) -------------
__global__ __launch_bounds__(256)
void tilesum_kernel(const float* __restrict__ inT, const float* __restrict__ inN,
                    float* __restrict__ T)
{
    __shared__ float red[256];
    const int tid = threadIdx.x;
    int bid = blockIdx.x;                     // ((n*6+c)*8+ti)*8+tj, 3072 total
    const int tj = bid & 7;  bid >>= 3;
    const int ti = bid & 7;  bid >>= 3;
    const int c  = bid % 6;
    const int n  = bid / 6;

    const float* src = (c < 3) ? (inT + (size_t)(n * 3 + c) * PLANE)
                               : (inN + (size_t)(n * 3 + (c - 3)) * PLANE);
    const size_t base = (size_t)(ti * 64) * HW + (size_t)(tj * 64);

    float s = 0.0f;
#pragma unroll
    for (int k = 0; k < 16; ++k) {
        int e = tid + k * 256;                // 64x64 tile, coalesced 64-wide rows
        s += src[base + (size_t)(e >> 6) * HW + (e & 63)];
    }
    red[tid] = s;
    __syncthreads();
    for (int off = 128; off > 0; off >>= 1) {
        if (tid < off) red[tid] += red[tid + off];
        __syncthreads();
    }
    if (tid == 0) T[blockIdx.x] = red[0];
}

// ---- lighting (27x6 mix of pooled means) + analytic zero overlap loss ------
__global__ void lighting_kernel(const float* __restrict__ T,
                                const float* __restrict__ Wl, const float* __restrict__ bl,
                                float* __restrict__ out)
{
    const int t = threadIdx.x;
    if (t == 0) out[LOSS_IDX] = 0.0f;  // overlap patches see identical pixels -> loss == 0
    if (t >= 392) return;
    const int n = t / 49;
    const int r = t % 49;
    const int i = r / 7, j = r % 7;

    float pooled[6];
#pragma unroll
    for (int c = 0; c < 6; ++c) {
        const float* Tc = T + (size_t)((n * 6 + c) * 64);
        pooled[c] = (Tc[i * 8 + j] + Tc[(i + 1) * 8 + j] +
                     Tc[i * 8 + j + 1] + Tc[(i + 1) * 8 + j + 1]) * (1.0f / 16384.0f);
    }
    for (int o = 0; o < 27; ++o) {
        float v = bl[o];
#pragma unroll
        for (int c = 0; c < 6; ++c) v += Wl[o * 6 + c] * pooled[c];
        out[LIGHT_OFF + (size_t)((n * 27 + o) * 49 + i * 7 + j)] = v;
    }
}

extern "C" void kernel_launch(void* const* d_in, const int* in_sizes, int n_in,
                              void* d_out, int out_size, void* d_ws, size_t ws_size,
                              hipStream_t stream) {
    const float* inT = (const float*)d_in[0];
    const float* inN = (const float*)d_in[1];
    const float* Wa  = (const float*)d_in[2];
    const float* ba  = (const float*)d_in[3];
    const float* Wn  = (const float*)d_in[4];
    const float* bn  = (const float*)d_in[5];
    const float* Ws  = (const float*)d_in[6];
    const float* bs  = (const float*)d_in[7];
    const float* Wl  = (const float*)d_in[8];
    const float* bl  = (const float*)d_in[9];
    float* out = (float*)d_out;
    float* T   = (float*)d_ws;                 // 3072 floats of scratch

    tilesum_kernel<<<dim3(3072), dim3(256), 0, stream>>>(inT, inN, T);
    lighting_kernel<<<dim3(1), dim3(512), 0, stream>>>(T, Wl, bl, out);
    heads_wmma_kernel<<<dim3(4096), dim3(256), 0, stream>>>(
        inT, inN, Wa, ba, Wn, bn, Ws, bs, out);
}
